// AdjacencyMatrix_91079076479328
// MI455X (gfx1250) — matrix operand order repaired and checked by reading, not compiled, not run on hardware
//
#include <hip/hip_runtime.h>

typedef __attribute__((ext_vector_type(16))) _Float16 v16h;
typedef __attribute__((ext_vector_type(8)))  _Float16 v8h;
typedef __attribute__((ext_vector_type(8)))  float    v8f;
typedef __attribute__((ext_vector_type(4)))  float    v4f;

#define N_NEURONS   1024
#define IN_NEURONS  256
#define OUT_NEURONS 128
#define ENERGY_SCALAR 0.9f

#define WAVES_PER_BLOCK 4
#define ROWS_PER_WAVE   16
#define ROWS_PER_BLOCK  (WAVES_PER_BLOCK * ROWS_PER_WAVE)
#define N_TILES         (N_NEURONS / 16)   // 64 column tiles
#define NT_CHUNK        8                  // tiles per accumulator chunk
#define K_BLOCK         256                // K elements per unrolled block (8 WMMA k-steps)

// ---------------------------------------------------------------------------
// One-time: W (f32 row-major, [k][n]) -> Wt (f16, transposed: Wt[n][k]).
// Transposed layout makes each WMMA B-fragment a contiguous 32B per-lane load.
// ---------------------------------------------------------------------------
__global__ void wt_convert_kernel(const float* __restrict__ W,
                                  _Float16* __restrict__ Wt) {
    const int n = blockIdx.x;  // column of W == row of Wt
    for (int k = threadIdx.x; k < N_NEURONS; k += blockDim.x) {
        Wt[(size_t)n * N_NEURONS + k] = (_Float16)W[(size_t)k * N_NEURONS + n];
    }
}

// ---------------------------------------------------------------------------
// Main kernel: each wave owns a 16-row stripe of m, double-buffered in LDS,
// runs num_steps of m <- 0.9*(m @ W) with v_wmma_f32_16x16x32_f16, then the
// diagonal-scaled tail extraction.
// ---------------------------------------------------------------------------
__launch_bounds__(WAVES_PER_BLOCK * 32)
__global__ void energy_steps_kernel(const float* __restrict__ x,
                                    const float* __restrict__ W,
                                    const _Float16* __restrict__ Wt,
                                    const int* __restrict__ steps_p,
                                    float* __restrict__ out) {
    // 2 buffers * 4 waves * 16 rows * 1024 cols * 2B = 256 KB (<= 320 KB WGP LDS)
    __shared__ __align__(16) _Float16 mbuf[2][WAVES_PER_BLOCK][ROWS_PER_WAVE][N_NEURONS];

    const int lane = threadIdx.x & 31;
    const int wave = threadIdx.x >> 5;
    const int row0 = blockIdx.x * ROWS_PER_BLOCK + wave * ROWS_PER_WAVE;
    const int ln   = lane & 15;   // M (A-frag) / N (B-frag) position within tile
    const int hi   = lane >> 4;   // lane half selects K sub-range per ISA layout
    const int num_steps = *steps_p;

    // ---- m0 = pad(x): cols [0,256) from x (f32->f16), cols [256,1024) zero ----
    for (int r = 0; r < ROWS_PER_WAVE; ++r) {
        const float* xr = x + (size_t)(row0 + r) * IN_NEURONS;
        {
            const int c = lane * 8;            // 32 lanes * 8 = 256 cols exactly
            v4f a = *(const v4f*)(xr + c);
            v4f b = *(const v4f*)(xr + c + 4);
            v8h h;
#pragma unroll
            for (int q = 0; q < 4; ++q) {
                h[q]     = (_Float16)a[q];
                h[4 + q] = (_Float16)b[q];
            }
            *(v8h*)&mbuf[0][wave][r][c] = h;
        }
        const v8h z = {};
        for (int c = IN_NEURONS + lane * 8; c < N_NEURONS; c += 32 * 8) {
            *(v8h*)&mbuf[0][wave][r][c] = z;
        }
    }

    // ---- num_steps of m <- 0.9 * (m @ W) ----
    for (int s = 0; s < num_steps; ++s) {
        const int rb = s & 1;       // read buffer
        const int wb = rb ^ 1;      // write buffer
        // Step 0: columns >= 256 of m0 are zero -> skip those K chunks.
        const int kblocks = (s == 0) ? (IN_NEURONS / K_BLOCK) : (N_NEURONS / K_BLOCK);

        for (int nt0 = 0; nt0 < N_TILES; nt0 += NT_CHUNK) {
            v8f acc[NT_CHUNK];
#pragma unroll
            for (int j = 0; j < NT_CHUNK; ++j) acc[j] = (v8f){};

            // Loop-invariant per-lane base pointers: K advances become
            // constant immediate offsets inside the unrolled block.
            const _Float16* abase = &mbuf[rb][wave][ln][hi * 16];
            const _Float16* bbase[NT_CHUNK];
#pragma unroll
            for (int j = 0; j < NT_CHUNK; ++j) {
                bbase[j] = Wt + (size_t)((nt0 + j) * 16 + ln) * N_NEURONS + hi * 16;
            }

            for (int kb0 = 0; kb0 < kblocks; ++kb0) {
                const int k0 = kb0 * K_BLOCK;
#pragma unroll
                for (int kk = 0; kk < K_BLOCK; kk += 32) {
                    // A fragment (16x32 f16): lane<16 holds K 0..15 of
                    // row ln, lane>=16 holds K 16..31.
                    v8h alo = *(const v8h*)(abase + k0 + kk);
                    v8h ahi = *(const v8h*)(abase + k0 + kk + 8);
                    v16h afrag = __builtin_shufflevector(alo, ahi,
                        0, 1, 2, 3, 4, 5, 6, 7, 8, 9, 10, 11, 12, 13, 14, 15);

                    // B fragment (32x16 f16): lane<16 holds K 0..15 of column
                    // n, lane>=16 holds K 16..31 -> contiguous 32B in Wt[n][k].
#pragma unroll
                    for (int j = 0; j < NT_CHUNK; ++j) {
                        v16h bfrag = *(const v16h*)(bbase[j] + k0 + kk);
                        acc[j] = __builtin_amdgcn_wmma_f32_16x16x32_f16(
                            /*neg_a=*/false, afrag, /*neg_b=*/false, bfrag,
                            /*c_mod=*/(short)0, acc[j],
                            /*reuse_a=*/false, /*reuse_b=*/false);
                    }
                }
            }

            // Writeback: D VGPR r -> row (hi*8 + r), col = tile*16 + ln.
            const int mrow = hi * 8;
#pragma unroll
            for (int j = 0; j < NT_CHUNK; ++j) {
                const int n = (nt0 + j) * 16 + ln;
#pragma unroll
                for (int r = 0; r < 8; ++r) {
                    mbuf[wb][wave][mrow + r][n] =
                        (_Float16)(ENERGY_SCALAR * acc[j][r]);
                }
            }
        }
    }

    // ---- y[b,c] = m_final[b, 896+c] * W[896+c][896+c] ----
    const int fb = num_steps & 1;
    for (int r = 0; r < ROWS_PER_WAVE; ++r) {
        const int c = lane * 4;                       // 32 lanes * 4 = 128 cols
#pragma unroll
        for (int q = 0; q < 4; ++q) {
            const int d = (N_NEURONS - OUT_NEURONS) + c + q;
            out[(size_t)(row0 + r) * OUT_NEURONS + c + q] =
                (float)mbuf[fb][wave][r][d] * W[(size_t)d * N_NEURONS + d];
        }
    }
}

// ---------------------------------------------------------------------------
// d_in[0] = x [16384,256] f32, d_in[1] = weight [1024,1024] f32,
// d_in[2] = num_steps (int, 1 elem). d_out = y [16384,128] f32.
// d_ws: 2 MB used for the f16 transposed weight.
// ---------------------------------------------------------------------------
extern "C" void kernel_launch(void* const* d_in, const int* in_sizes, int n_in,
                              void* d_out, int out_size, void* d_ws, size_t ws_size,
                              hipStream_t stream) {
    const float* x     = (const float*)d_in[0];
    const float* W     = (const float*)d_in[1];
    const int*   steps = (const int*)d_in[2];
    float*       out   = (float*)d_out;
    _Float16*    Wt    = (_Float16*)d_ws;   // 1024*1024*2 = 2 MB

    wt_convert_kernel<<<N_NEURONS, 256, 0, stream>>>(W, Wt);

    const int nrows = in_sizes[0] / IN_NEURONS;  // 16384
    energy_steps_kernel<<<nrows / ROWS_PER_BLOCK, WAVES_PER_BLOCK * 32, 0, stream>>>(
        x, W, Wt, steps, out);
}
